// MulticlassDice_59442347376955
// MI455X (gfx1250) — 8-bit matrix operand order repaired and checked by reading, compiled, not run on hardware
//
#include <hip/hip_runtime.h>
#include <hip/hip_bf16.h>


#define NCLS 8
#define NSAMP 8
#define PIX (512 * 512)
#define BLOCKS_PER_SAMPLE 16
#define WAVES_PER_BLOCK 8
#define WAVES_PER_SAMPLE (BLOCKS_PER_SAMPLE * WAVES_PER_BLOCK)
#define PIX_PER_WAVE (PIX / WAVES_PER_SAMPLE)
#define ITERS (PIX_PER_WAVE / 128)

typedef __attribute__((ext_vector_type(8))) int v8i;
typedef unsigned int uint_t;

__global__ void __launch_bounds__(256)
dice_confusion_kernel(const int* __restrict__ in, const int* __restrict__ tg,
                      unsigned int* __restrict__ conf) {
    __shared__ unsigned int lin[WAVES_PER_BLOCK][32];
    __shared__ unsigned int ltg[WAVES_PER_BLOCK][32];

    const int lane = threadIdx.x & 31;
    const int wv   = threadIdx.x >> 5;
    const int s    = blockIdx.x >> 4;
    const int wis  = (blockIdx.x & 15) * WAVES_PER_BLOCK + wv;
    const int base = s * PIX + wis * PIX_PER_WAVE;

    const unsigned int m   = (unsigned int)(lane & 15);
    const unsigned int lo  = (m < 4u) ? (1u << (8u * m)) : 0u;
    const unsigned int hi  = (m >= 4u && m < 8u) ? (1u << (8u * (m - 4u))) : 0u;
    const int khalf = (lane >> 4) * 8;
    const int boff  = (lane >> 4) * 16;

    v8i acc = {};

    for (int it = 0; it < ITERS; ++it) {
        const int p = base + it * 128;

        const int4 iv = ((const int4*)(in + p))[lane];
        const int4 tv = ((const int4*)(tg + p))[lane];

        __builtin_prefetch(in + p + 512, 0, 0);
        __builtin_prefetch(tg + p + 512, 0, 0);

        const unsigned int pin = (unsigned int)iv.x | ((unsigned int)iv.y << 8) |
                                 ((unsigned int)iv.z << 16) | ((unsigned int)iv.w << 24);
        const unsigned int ptg = (unsigned int)tv.x | ((unsigned int)tv.y << 8) |
                                 ((unsigned int)tv.z << 16) | ((unsigned int)tv.w << 24);
        lin[wv][lane] = pin;
        ltg[wv][lane] = ptg;
        __syncthreads();

#pragma unroll
        for (int c = 0; c < 2; ++c) {
            const char* cin = (const char*)&lin[wv][c * 16];
            const char* ctg = (const char*)&ltg[wv][c * 16];

            const uint2 a01 = *(const uint2*)(cin + 2 * khalf);
            const uint2 a23 = *(const uint2*)(cin + 8 + 2 * khalf);
            const uint2 a45 = *(const uint2*)(cin + 32 + 2 * khalf);
            const uint2 a67 = *(const uint2*)(cin + 40 + 2 * khalf);
            v8i a;
            a[0] = (int)__builtin_amdgcn_perm(hi, lo, a01.x);
            a[1] = (int)__builtin_amdgcn_perm(hi, lo, a01.y);
            a[2] = (int)__builtin_amdgcn_perm(hi, lo, a23.x);
            a[3] = (int)__builtin_amdgcn_perm(hi, lo, a23.y);
            a[4] = (int)__builtin_amdgcn_perm(hi, lo, a45.x);
            a[5] = (int)__builtin_amdgcn_perm(hi, lo, a45.y);
            a[6] = (int)__builtin_amdgcn_perm(hi, lo, a67.x);
            a[7] = (int)__builtin_amdgcn_perm(hi, lo, a67.y);

            const uint4 b03 = *(const uint4*)(ctg + boff);
            const uint4 b47 = *(const uint4*)(ctg + boff + 32);
            v8i b;
            b[0] = (int)__builtin_amdgcn_perm(hi, lo, b03.x);
            b[1] = (int)__builtin_amdgcn_perm(hi, lo, b03.y);
            b[2] = (int)__builtin_amdgcn_perm(hi, lo, b03.z);
            b[3] = (int)__builtin_amdgcn_perm(hi, lo, b03.w);
            b[4] = (int)__builtin_amdgcn_perm(hi, lo, b47.x);
            b[5] = (int)__builtin_amdgcn_perm(hi, lo, b47.y);
            b[6] = (int)__builtin_amdgcn_perm(hi, lo, b47.z);
            b[7] = (int)__builtin_amdgcn_perm(hi, lo, b47.w);

            acc = __builtin_amdgcn_wmma_i32_16x16x64_iu8(false, a, false, b, acc,
                                                         false, false);
        }
        __syncthreads();
    }

    const int ncol = lane & 15;
    if ((lane >> 4) == 0 && ncol < NCLS) {
#pragma unroll
        for (int r = 0; r < NCLS; ++r) {
            atomicAdd(&conf[s * (NCLS * NCLS) + r * NCLS + ncol],
                      (unsigned int)acc[r]);
        }
    }
}

__global__ void zero_conf_kernel(unsigned int* __restrict__ conf) {
    conf[threadIdx.x] = 0u;
}

__global__ void dice_reduce_kernel(const unsigned int* __restrict__ conf,
                                   const float* __restrict__ w,
                                   float* __restrict__ out) {
    __shared__ float dice_s[NSAMP][NCLS];
    __shared__ float dcls_s[NCLS];
    const int t = threadIdx.x;
    const int n = t >> 3, c = t & 7;

    const unsigned int* cn = conf + n * (NCLS * NCLS);
    unsigned int inter = cn[c * NCLS + c];
    unsigned int rs = 0, cs = 0;
#pragma unroll
    for (int k = 0; k < NCLS; ++k) {
        rs += cn[c * NCLS + k];
        cs += cn[k * NCLS + c];
    }
    const float SM = 1e-5f;
    dice_s[n][c] = (2.0f * (float)inter + SM) / ((float)(rs + cs) + SM);
    __syncthreads();

    if (t < NCLS) {
        float acc = 0.0f;
#pragma unroll
        for (int nn = 0; nn < NSAMP; ++nn) acc += dice_s[nn][t];
        const float d = acc * (1.0f / (float)NSAMP);
        dcls_s[t] = d;
        out[1 + t] = d;
    }
    __syncthreads();

    if (t == 0) {
        float tot = 0.0f;
#pragma unroll
        for (int cc = 0; cc < NCLS; ++cc) tot += w[cc] * dcls_s[cc];
        out[0] = tot;
    }
}

extern "C" void kernel_launch(void* const* d_in, const int* in_sizes, int n_in,
                              void* d_out, int out_size, void* d_ws, size_t ws_size,
                              hipStream_t stream) {
    (void)in_sizes; (void)n_in; (void)out_size; (void)ws_size;
    const int*   in  = (const int*)d_in[0];
    const int*   tg  = (const int*)d_in[1];
    const float* wts = (const float*)d_in[2];
    float*       out = (float*)d_out;
    unsigned int* conf = (unsigned int*)d_ws;

    zero_conf_kernel<<<1, NSAMP * NCLS * NCLS, 0, stream>>>(conf);
    dice_confusion_kernel<<<NSAMP * BLOCKS_PER_SAMPLE, 32 * WAVES_PER_BLOCK, 0, stream>>>(
        in, tg, conf);
    dice_reduce_kernel<<<1, NSAMP * NCLS, 0, stream>>>(conf, wts, out);
}
